// AlgebraicAttention_6614249636074
// MI455X (gfx1250) — matrix operand order repaired and checked by reading, not compiled, not run on hardware
//
#include <hip/hip_runtime.h>
#include <hip/hip_bf16.h>

typedef __attribute__((ext_vector_type(16))) _Float16 v16h;
typedef __attribute__((ext_vector_type(8)))  _Float16 v8h;
typedef __attribute__((ext_vector_type(4)))  _Float16 v4h;
typedef __attribute__((ext_vector_type(2)))  _Float16 v2h;
typedef __attribute__((ext_vector_type(2)))  __fp16   v2hp;   // pkrtz return type
typedef __attribute__((ext_vector_type(8)))  float    v8f;
typedef __attribute__((ext_vector_type(4)))  float    v4f;

#define D_MODEL 1024
#define N_HEAD  16
#define DH      64
#define T_SEQ   2048
#define BATCH   2
#define EPS_F   1e-6f

// LDS row pads (halves): keep 16-byte row alignment, break 64-bank alignment.
#define AROW (32 + 8)   // 80-byte rows
#define BROW (32 + 8)   // 80-byte rows
#define KROW (64 + 8)   // 144-byte rows

static __device__ __forceinline__ v2h pk2(float a, float b) {
    union { v2hp p; v2h h; } u;
    u.p = __builtin_amdgcn_cvt_pkrtz(a, b);
    return u.h;
}

static __device__ __forceinline__ v4h cvt4(const v4f f) {
    union { v2h h2[2]; v4h h4; } u;
    u.h2[0] = pk2(f[0], f[1]);
    u.h2[1] = pk2(f[2], f[3]);
    return u.h4;
}

// ---------------------------------------------------------------------------
// Fragment layouts used here (CDNA5 wave32 WMMA; A/B k order contiguous, not the ISA table order):
//   A 16x32 f16 : lane m = lane&15, hi = lane>>4 ; elem e -> K = e + hi*16
//   B 32x16 f16 : lane n = lane&15, hi = lane>>4 ; elem e -> K = e + hi*16
//                 (B tiles staged fragment-major: [n][k], contiguous per lane)
//   C/D 16x16 f32: 8 VGPRs, elem r -> M = r + hi*8 ; N = lane&15
// ---------------------------------------------------------------------------

// ===========================================================================
// Kernel 1: fused Q/K/V projection.  Y[m,n] = sum_k X[m,k] * W[n,k]
// Block: 128 threads (4 waves), tile 128(M) x 64(N), BK=32.
// Q,K -> [B,H,T,Dh] f16 ;  V -> [B,H,Dh,T] f16 (pre-transposed for PV B-frags)
// ===========================================================================
__global__ __launch_bounds__(128)
void aa_qkv_proj_kernel(const float* __restrict__ x,
                        const float* __restrict__ Wq,
                        const float* __restrict__ Wk,
                        const float* __restrict__ Wv,
                        _Float16* __restrict__ qws,
                        _Float16* __restrict__ kws,
                        _Float16* __restrict__ vws)
{
    const int z = blockIdx.z;
    const float* __restrict__ W = (z == 0) ? Wq : (z == 1) ? Wk : Wv;
    _Float16* __restrict__ outp = (z == 0) ? qws : (z == 1) ? kws : vws;

    const int nBase = blockIdx.x * 64;
    const int mBase = blockIdx.y * 128;
    const int tid  = threadIdx.x;
    const int lane = tid & 31;
    const int wv   = tid >> 5;     // 0..3
    const int lm   = lane & 15;
    const int hi   = lane >> 4;

    __shared__ _Float16 As[128 * AROW];   // [m][k]
    __shared__ _Float16 Bs[64 * BROW];    // [n][k]  (fragment-major)

    // per-thread staging bases: chunk i at constant offset i*16*D_MODEL
    const int rS = tid >> 3, cS = (tid & 7) * 4;
    const float* pA = x + (size_t)(mBase + rS) * D_MODEL + cS;
    const float* pB = W + (size_t)(nBase + rS) * D_MODEL + cS;
    _Float16* sA = &As[rS * AROW + cS];
    _Float16* sB = &Bs[rS * BROW + cS];

    v8f acc[2][4] = {};

    for (int k0 = 0; k0 < D_MODEL; k0 += 32) {
        // ---- phase 1: clause global loads (constant chunk offsets) ----
        v4f ta[8], tb[4];
        #pragma unroll
        for (int i = 0; i < 8; ++i)
            ta[i] = *(const v4f*)(pA + (size_t)i * 16 * D_MODEL);
        #pragma unroll
        for (int i = 0; i < 4; ++i)
            tb[i] = *(const v4f*)(pB + (size_t)i * 16 * D_MODEL);
        if (k0 + 32 < D_MODEL) {
            __builtin_prefetch(pA + 32, 0, 1);
            __builtin_prefetch(pB + 32, 0, 1);
        }
        pA += 32;
        pB += 32;
        __syncthreads();
        // ---- phase 2: packed converts + wide LDS stores ----
        #pragma unroll
        for (int i = 0; i < 8; ++i)
            *(v4h*)(sA + i * 16 * AROW) = cvt4(ta[i]);
        #pragma unroll
        for (int i = 0; i < 4; ++i)
            *(v4h*)(sB + i * 16 * BROW) = cvt4(tb[i]);
        __syncthreads();

        // ---- compute: 8 WMMA per K-step ----
        v16h a[2];
        #pragma unroll
        for (int st = 0; st < 2; ++st) {
            const int m0 = (wv + st * 4) * 16;
            #pragma unroll
            for (int e = 0; e < 16; ++e) {
                int kk = hi * 16 + e;
                a[st][e] = As[(m0 + lm) * AROW + kk];
            }
        }
        #pragma unroll
        for (int nt = 0; nt < 4; ++nt) {
            v16h b;
            #pragma unroll
            for (int e = 0; e < 16; ++e)
                b[e] = Bs[(nt * 16 + lm) * BROW + hi * 16 + e];
            acc[0][nt] = __builtin_amdgcn_wmma_f32_16x16x32_f16(
                false, a[0], false, b, (short)0, acc[0][nt], false, false);
            acc[1][nt] = __builtin_amdgcn_wmma_f32_16x16x32_f16(
                false, a[1], false, b, (short)0, acc[1][nt], false, false);
        }
    }

    // ---- writeback ----
    #pragma unroll
    for (int st = 0; st < 2; ++st) {
        #pragma unroll
        for (int nt = 0; nt < 4; ++nt) {
            int col = nBase + nt * 16 + lm;      // c_out
            int h = col >> 6, d = col & 63;
            int rowb = mBase + (wv + st * 4) * 16 + hi * 8;   // b*T + t base
            int b = rowb >> 11, tb0 = rowb & (T_SEQ - 1);
            size_t hb = ((size_t)b * N_HEAD + h) * (size_t)T_SEQ * DH;
            if (z == 2) {
                // V: [B,H,Dh,T] -> t-contiguous: pack pairs, 4B stores
                #pragma unroll
                for (int r = 0; r < 4; ++r) {
                    *(v2h*)&outp[hb + (size_t)d * T_SEQ + tb0 + 2 * r] =
                        pk2(acc[st][nt][2 * r], acc[st][nt][2 * r + 1]);
                }
            } else {
                #pragma unroll
                for (int r = 0; r < 8; ++r)
                    outp[hb + (size_t)(tb0 + r) * DH + d] = (_Float16)acc[st][nt][r];
            }
        }
    }
}

// ===========================================================================
// Kernel 2: streaming attention with rational-power normalization.
// p = ((s/(|s|+1)+1)/2)^power in (0,1]: plain running sums, no rescaling.
// Causal: key tiles beyond the q-block skipped.
// ===========================================================================
__global__ __launch_bounds__(128)
void aa_attn_kernel(const _Float16* __restrict__ qws,
                    const _Float16* __restrict__ kws,
                    const _Float16* __restrict__ vws,
                    const float* __restrict__ alibi,
                    const float* __restrict__ powerp,
                    _Float16* __restrict__ attws)
{
    const int bh = blockIdx.y;              // b*H + h
    const int h  = bh & (N_HEAD - 1);
    const int b  = bh >> 4;
    const int qblock = blockIdx.x * 64;
    const int tid  = threadIdx.x;
    const int lane = tid & 31;
    const int wv   = tid >> 5;
    const int lm   = lane & 15;
    const int hi   = lane >> 4;

    const float slope = alibi[h];
    const float power = powerp[0];
    const float scale = 2.0f;               // (1/sqrt(64)) * 16

    const size_t headBase = (size_t)bh * T_SEQ * DH;

    __shared__ _Float16 Ks[32 * KROW];      // [key][d]  (d contiguous)
    __shared__ _Float16 Vt[DH * BROW];      // [d][key]  (fragment-major)
    __shared__ _Float16 Ps[4][16 * AROW];   // per-wave P [m][k]

    // Preload Q A-fragments for both Dh chunks (live in VGPRs all loop)
    v16h qa[2];
    const int qrow = qblock + wv * 16 + lm;
    #pragma unroll
    for (int dc = 0; dc < 2; ++dc)
        #pragma unroll
        for (int e = 0; e < 16; ++e) {
            int kk = hi * 16 + e;
            qa[dc][e] = qws[headBase + (size_t)qrow * DH + dc * 32 + kk];
        }

    // staging bases (advance per key tile)
    const int rK = tid >> 3, cK = (tid & 7) * 8;       // K: [key][d]
    const int rV = tid >> 2, cV = (tid & 3) * 8;       // V: [d][key]
    const _Float16* pK = kws + headBase + (size_t)rK * DH + cK;
    const _Float16* pV = vws + headBase + (size_t)rV * T_SEQ + cV;
    _Float16* sK = &Ks[rK * KROW + cK];
    _Float16* sV = &Vt[rV * BROW + cV];

    v8f accO[4] = {};
    float dsum[8];
    #pragma unroll
    for (int r = 0; r < 8; ++r) dsum[r] = 0.0f;

    const int kEnd = qblock + 64;           // causal upper bound
    for (int kbase = 0; kbase < kEnd; kbase += 32) {
        // ---- stage K[32][64] and Vt[64][32], 8-half chunks ----
        v8h tk[2], tv[2];
        #pragma unroll
        for (int i = 0; i < 2; ++i) {
            tk[i] = *(const v8h*)(pK + (size_t)i * 16 * DH);
            tv[i] = *(const v8h*)(pV + (size_t)i * 32 * T_SEQ);
        }
        if (kbase + 32 < kEnd) {
            __builtin_prefetch(pK + 32 * DH, 0, 1);
            __builtin_prefetch(pV + 32, 0, 1);
        }
        pK += 32 * DH;
        pV += 32;
        __syncthreads();
        #pragma unroll
        for (int i = 0; i < 2; ++i) {
            *(v8h*)(sK + i * 16 * KROW) = tk[i];
            *(v8h*)(sV + i * 32 * BROW) = tv[i];
        }
        __syncthreads();

        // ---- S = Q . K^T -> two 16x16 tiles, then p(s) ----
        #pragma unroll
        for (int nt = 0; nt < 2; ++nt) {
            v8f accS = {};
            #pragma unroll
            for (int dc = 0; dc < 2; ++dc) {
                v16h kb;
                #pragma unroll
                for (int e = 0; e < 16; ++e)
                    kb[e] = Ks[(nt * 16 + lm) * KROW + dc * 32 + hi * 16 + e];
                accS = __builtin_amdgcn_wmma_f32_16x16x32_f16(
                    false, qa[dc], false, kb, (short)0, accS, false, false);
            }
            const int kidx = kbase + nt * 16 + lm;
            #pragma unroll
            for (int r = 0; r < 8; ++r) {
                int qi = qblock + wv * 16 + r + hi * 8;
                float s = accS[r] * scale;
                float dist = (float)(kidx - qi);
                s -= slope * __builtin_fabsf(dist);
                if (kidx > qi) s = -10000.0f;
                float t = s / (__builtin_fabsf(s) + 1.0f);
                float u = (t + 1.0f) * 0.5f;                 // (0,1]
                float p = __builtin_exp2f(power * __builtin_log2f(u));
                dsum[r] += p;
                Ps[wv][(r + hi * 8) * AROW + nt * 16 + lm] = (_Float16)p;
            }
        }
        __syncthreads();   // Ps re-striping across lanes

        // ---- O += P (16x32) . V (32x64) ----
        v16h pa;
        #pragma unroll
        for (int e = 0; e < 16; ++e) {
            int kk = hi * 16 + e;
            pa[e] = Ps[wv][lm * AROW + kk];
        }
        #pragma unroll
        for (int dt = 0; dt < 4; ++dt) {
            v16h vb;
            #pragma unroll
            for (int e = 0; e < 16; ++e)
                vb[e] = Vt[(dt * 16 + lm) * BROW + hi * 16 + e];
            accO[dt] = __builtin_amdgcn_wmma_f32_16x16x32_f16(
                false, pa, false, vb, (short)0, accO[dt], false, false);
        }
    }

    // Row sums: reduce across the 16 lanes of each half-wave
    #pragma unroll
    for (int r = 0; r < 8; ++r) {
        float v = dsum[r];
        #pragma unroll
        for (int m = 1; m < 16; m <<= 1)
            v += __shfl_xor(v, m, 32);
        dsum[r] = v + EPS_F;
    }

    // Normalize and write attn output (f16) into [B, T, C]
    #pragma unroll
    for (int dt = 0; dt < 4; ++dt) {
        int c = h * DH + dt * 16 + lm;
        #pragma unroll
        for (int r = 0; r < 8; ++r) {
            int t = qblock + wv * 16 + r + hi * 8;
            attws[((size_t)b * T_SEQ + t) * D_MODEL + c] =
                (_Float16)(accO[dt][r] / dsum[r]);
        }
    }
}

// ===========================================================================
// Kernel 3: output projection.  out[m,n] = sum_k att[m,k] * Wo[n,k] (f32 out)
// ===========================================================================
__global__ __launch_bounds__(128)
void aa_out_proj_kernel(const _Float16* __restrict__ att,
                        const float* __restrict__ Wo,
                        float* __restrict__ out)
{
    const int nBase = blockIdx.x * 64;
    const int mBase = blockIdx.y * 128;
    const int tid  = threadIdx.x;
    const int lane = tid & 31;
    const int wv   = tid >> 5;
    const int lm   = lane & 15;
    const int hi   = lane >> 4;

    __shared__ _Float16 As[128 * AROW];
    __shared__ _Float16 Bs[64 * BROW];

    const int rA = tid >> 2, cA = (tid & 3) * 8;       // A (f16): 8-half chunks
    const int rB = tid >> 3, cB = (tid & 7) * 4;       // B (f32): float4 chunks
    const _Float16* pA = att + (size_t)(mBase + rA) * D_MODEL + cA;
    const float*    pB = Wo  + (size_t)(nBase + rB) * D_MODEL + cB;
    _Float16* sA = &As[rA * AROW + cA];
    _Float16* sB = &Bs[rB * BROW + cB];

    v8f acc[2][4] = {};

    for (int k0 = 0; k0 < D_MODEL; k0 += 32) {
        v8h ta[4];
        v4f tb[4];
        #pragma unroll
        for (int i = 0; i < 4; ++i)
            ta[i] = *(const v8h*)(pA + (size_t)i * 32 * D_MODEL);
        #pragma unroll
        for (int i = 0; i < 4; ++i)
            tb[i] = *(const v4f*)(pB + (size_t)i * 16 * D_MODEL);
        if (k0 + 32 < D_MODEL) {
            __builtin_prefetch(pA + 32, 0, 1);
            __builtin_prefetch(pB + 32, 0, 1);
        }
        pA += 32;
        pB += 32;
        __syncthreads();
        #pragma unroll
        for (int i = 0; i < 4; ++i)
            *(v8h*)(sA + i * 32 * AROW) = ta[i];
        #pragma unroll
        for (int i = 0; i < 4; ++i)
            *(v4h*)(sB + i * 16 * BROW) = cvt4(tb[i]);
        __syncthreads();

        v16h a[2];
        #pragma unroll
        for (int st = 0; st < 2; ++st) {
            const int m0 = (wv + st * 4) * 16;
            #pragma unroll
            for (int e = 0; e < 16; ++e) {
                int kk = hi * 16 + e;
                a[st][e] = As[(m0 + lm) * AROW + kk];
            }
        }
        #pragma unroll
        for (int nt = 0; nt < 4; ++nt) {
            v16h bfr;
            #pragma unroll
            for (int e = 0; e < 16; ++e)
                bfr[e] = Bs[(nt * 16 + lm) * BROW + hi * 16 + e];
            acc[0][nt] = __builtin_amdgcn_wmma_f32_16x16x32_f16(
                false, a[0], false, bfr, (short)0, acc[0][nt], false, false);
            acc[1][nt] = __builtin_amdgcn_wmma_f32_16x16x32_f16(
                false, a[1], false, bfr, (short)0, acc[1][nt], false, false);
        }
    }

    #pragma unroll
    for (int st = 0; st < 2; ++st) {
        #pragma unroll
        for (int nt = 0; nt < 4; ++nt) {
            int col = nBase + nt * 16 + lm;
            #pragma unroll
            for (int r = 0; r < 8; ++r) {
                int row = mBase + (wv + st * 4) * 16 + r + hi * 8;
                out[(size_t)row * D_MODEL + col] = acc[st][nt][r];
            }
        }
    }
}

// ===========================================================================
extern "C" void kernel_launch(void* const* d_in, const int* in_sizes, int n_in,
                              void* d_out, int out_size, void* d_ws, size_t ws_size,
                              hipStream_t stream)
{
    const float* x     = (const float*)d_in[0];
    const float* Wq    = (const float*)d_in[1];
    const float* Wk    = (const float*)d_in[2];
    const float* Wv    = (const float*)d_in[3];
    const float* Wo    = (const float*)d_in[4];
    const float* alibi = (const float*)d_in[5];
    const float* power = (const float*)d_in[6];

    char* ws = (char*)d_ws;
    const size_t perBuf = (size_t)BATCH * N_HEAD * T_SEQ * DH * sizeof(_Float16); // 8 MB
    _Float16* qws   = (_Float16*)(ws);
    _Float16* kws   = (_Float16*)(ws + perBuf);
    _Float16* vws   = (_Float16*)(ws + 2 * perBuf);   // [B,H,Dh,T]
    _Float16* attws = (_Float16*)(ws + 3 * perBuf);

    dim3 gProj(D_MODEL / 64, (BATCH * T_SEQ) / 128, 3);
    aa_qkv_proj_kernel<<<gProj, 128, 0, stream>>>(x, Wq, Wk, Wv, qws, kws, vws);

    dim3 gAttn(T_SEQ / 64, BATCH * N_HEAD);
    aa_attn_kernel<<<gAttn, 128, 0, stream>>>(qws, kws, vws, alibi, power, attws);

    dim3 gOut(D_MODEL / 64, (BATCH * T_SEQ) / 128);
    aa_out_proj_kernel<<<gOut, 128, 0, stream>>>(attws, Wo, (float*)d_out);
}
